// NonLocal_21792664060441
// MI455X (gfx1250) — matrix operand order repaired and checked by reading, not compiled, not run on hardware
//
#include <hip/hip_runtime.h>
#include <hip/hip_bf16.h>

// Non-local attention block, fused flash-attention style for MI455X (gfx1250).
// B=8, C=64, CI=32, N=4096. All GEMMs via v_wmma_f32_16x16x32_f16 (K=32==CI).
//
// Key layout trick: compute S^T = phi^T(keys x CI) @ theta^T(CI x queries) so the
// C-fragment assigns ONE QUERY PER LANE (n = lane&15). Online-softmax state is
// then lane-scalar; all cross-lane traffic is xor-16 partner exchanges.
// f ([B,N,N], 512 MB) is never materialized.

typedef __attribute__((ext_vector_type(16))) _Float16 v16h;
typedef __attribute__((ext_vector_type(2)))  __fp16   v2fp;
typedef __attribute__((ext_vector_type(8)))  float    v8f;

union Frag16 { v16h h; uint32_t u[8]; };
union Pk2    { v2fp h; uint32_t u; };

static __device__ inline uint32_t pk2(float a, float b) {
    Pk2 t; t.h = __builtin_amdgcn_cvt_pkrtz(a, b);   // v_cvt_pk_f16_f32
    return t.u;
}

#define NB   8
#define NC   64
#define NCI  32
#define NN   4096   // H*W

// ---------------------------------------------------------------------------
// Stage 1: 1x1-conv projections -> f16 scratch, pre-swizzled into WMMA frag
// layouts so stage 2 fills every fragment with one 32B load per lane.
//   thetaB : B-frag of theta^T (k=ci, n=query):
//            [(b, q/16, lane=(ci>>4)*16|(q&15), h=ci&15)]
//   phiA   : A-frag of phi^T (r=key, k=ci):
//            [(b, key/16, lane=(ci>>4)*16|(key&15), h=ci&15)]
//   gTA    : A-frag of g^T (r=ci, k=key-in-32):
//            [(b, key/32, cihalf, lane=(kk>>4)*16|(ci&15), h=kk&15)]
// ---------------------------------------------------------------------------
__global__ __launch_bounds__(256)
void nl_proj_kernel(const float* __restrict__ x,
                    const float* __restrict__ w_theta, const float* __restrict__ b_theta,
                    const float* __restrict__ w_phi,   const float* __restrict__ b_phi,
                    const float* __restrict__ w_g,     const float* __restrict__ b_g,
                    _Float16* __restrict__ thetaB,
                    _Float16* __restrict__ phiA,
                    _Float16* __restrict__ gTA) {
    __shared__ float xs[NC][64 + 1];
    const int b  = blockIdx.y;
    const int n0 = blockIdx.x * 64;
    const int tid = threadIdx.x;

    for (int idx = tid; idx < NC * 64; idx += 256) {
        int c = idx >> 6, nn = idx & 63;
        xs[c][nn] = x[((size_t)b * NC + c) * NN + n0 + nn];   // coalesced in nn
    }
    __syncthreads();

    for (int p = 0; p < 8; ++p) {
        int q  = tid + p * 256;      // 0..2047 -> (nn, ci)
        int ci = q & 31;
        int nn = q >> 5;             // 0..63
        float th = b_theta[ci], ph = b_phi[ci], gg = b_g[ci];
        #pragma unroll
        for (int c = 0; c < NC; ++c) {
            float xv = xs[c][nn];
            th = fmaf(xv, w_theta[ci * NC + c], th);
            ph = fmaf(xv, w_phi  [ci * NC + c], ph);
            gg = fmaf(xv, w_g    [ci * NC + c], gg);
        }
        int n = n0 + nn;             // token index (query for theta, key for phi/g)
        {   // theta^T as B-fragment (k=ci, n=query)
            int j    = n >> 4;
            int lane = ((ci >> 4) << 4) | (n & 15);
            int h    = ci & 15;
            thetaB[(((size_t)b * (NN / 16) + j) * 32 + lane) * 16 + h] = (_Float16)th;
        }
        {   // phi^T as A-fragment (r=key, k=ci)
            int j    = n >> 4;
            int lane = ((ci >> 4) << 4) | (n & 15);
            int h    = ci & 15;
            phiA[(((size_t)b * (NN / 16) + j) * 32 + lane) * 16 + h] = (_Float16)ph;
        }
        {   // g^T as A-fragment (r=ci, k=key within 32-chunk)
            int mT     = n >> 5;
            int kk     = n & 31;
            int cihalf = ci >> 4;
            int lane   = ((kk >> 4) << 4) | (ci & 15);
            int h      = kk & 15;
            gTA[((((size_t)b * (NN / 32) + mT) * 2 + cihalf) * 32 + lane) * 16 + h] = (_Float16)gg;
        }
    }
}

// ---------------------------------------------------------------------------
// Stage 2: fused attention. 8 waves/WG, each wave owns 16 queries, fully
// wave-autonomous (no LDS tile, no block barriers). Per 64-key block:
// 4 WMMAs for S^T, lane-scalar online softmax, paired-lane xor-16 exchanges to
// form the P^T B-fragments, 4 WMMAs for y^T accumulation.
// ---------------------------------------------------------------------------
__global__ __launch_bounds__(256)
void nl_attn_kernel(const float* __restrict__ x,
                    const float* __restrict__ w_out, const float* __restrict__ b_out,
                    const _Float16* __restrict__ thetaB,
                    const _Float16* __restrict__ phiA,
                    const _Float16* __restrict__ gTA,
                    float* __restrict__ out) {
    const int b    = blockIdx.y;
    const int wave = threadIdx.x >> 5;
    const int lane = threadIdx.x & 31;
    const int hi   = lane >> 4;                // partner-half select
    const int ln   = lane & 15;                // this lane's query (within tile)
    const int qTile = blockIdx.x * 8 + wave;   // 16-query tile owned by this wave

    // theta^T B-fragment for our 16 queries: loaded once.
    const v16h bTheta = *(const v16h*)(thetaB + (((size_t)b * (NN / 16) + qTile) * 32 + lane) * 16);

    // w_out A-fragments (4 cout tiles of 16), loaded once: A[r=cout][k=ci].
    v16h aW[4];
    #pragma unroll
    for (int t = 0; t < 4; ++t) {
        const float* wrow = w_out + (t * 16 + ln) * NCI;
        #pragma unroll
        for (int h = 0; h < 16; ++h) {
            int ci = hi * 16 + h;
            aW[t][h] = (_Float16)wrow[ci];
        }
    }

    v8f acc0 = {}, acc1 = {};                  // y^T: rows ci 0-15 / 16-31, col = query ln
    float m = -1e30f, l = 0.0f;                // lane-scalar online-softmax state

    const _Float16* phiBase = phiA + (size_t)b * (NN / 16) * 32 * 16;
    const _Float16* gBase   = gTA  + (size_t)b * (NN / 32) * 2 * 32 * 16;

    for (int col = 0; col < NN; col += 64) {
        if (col + 64 < NN) {                   // global_prefetch_b8 for next key block
            __builtin_prefetch(phiBase + (((size_t)(col + 64) >> 4) * 32 + lane) * 16, 0, 1);
            __builtin_prefetch(gBase + ((((size_t)(col + 64) >> 5) * 2) * 32 + lane) * 16, 0, 1);
        }

        // S^T tiles: rows=keys (i + hi*8 within tile), col=query (ln).
        v8f s[4];
        #pragma unroll
        for (int t = 0; t < 4; ++t) {
            int keyTile = (col >> 4) + t;
            v16h aPhi = *(const v16h*)(phiBase + ((size_t)keyTile * 32 + lane) * 16);
            v8f z = {};
            s[t] = __builtin_amdgcn_wmma_f32_16x16x32_f16(false, aPhi, false, bTheta,
                                                          (short)0, z, false, false);
        }

        // Lane-scalar online softmax over the 64 keys of this block.
        float bm = s[0][0];
        #pragma unroll
        for (int t = 0; t < 4; ++t)
            #pragma unroll
            for (int i = 0; i < 8; ++i) bm = fmaxf(bm, s[t][i]);
        bm = fmaxf(bm, __shfl_xor(bm, 16, 32));       // combine partner half's keys
        float newm  = fmaxf(m, bm);
        float scale = __expf(m - newm);
        float ps = 0.0f;
        uint32_t pk[4][4];                            // exp'd P, f16-packed pairs
        #pragma unroll
        for (int t = 0; t < 4; ++t) {
            float p[8];
            #pragma unroll
            for (int i = 0; i < 8; ++i) { p[i] = __expf(s[t][i] - newm); ps += p[i]; }
            #pragma unroll
            for (int j = 0; j < 4; ++j) pk[t][j] = pk2(p[2 * j], p[2 * j + 1]);
        }
        ps += __shfl_xor(ps, 16, 32);
        l = l * scale + ps;
        m = newm;
        #pragma unroll
        for (int i = 0; i < 8; ++i) { acc0[i] *= scale; acc1[i] *= scale; }

        // y^T += g^T @ P^T over two 32-key chunks. P^T B-frag needs a single
        // partner exchange: each lane sends the tile its xor-16 partner needs.
        #pragma unroll
        for (int c = 0; c < 2; ++c) {
            uint32_t recv[4];
            #pragma unroll
            for (int j = 0; j < 4; ++j) {
                uint32_t send = hi ? pk[2 * c][j] : pk[2 * c + 1][j];
                recv[j] = (uint32_t)__shfl_xor((int)send, 16, 32);
            }
            Frag16 Bp;
            #pragma unroll
            for (int j = 0; j < 4; ++j) {
                Bp.u[j]     = hi ? recv[j]          : pk[2 * c][j];
                Bp.u[4 + j] = hi ? pk[2 * c + 1][j] : recv[j];
            }
            int mT = (col >> 5) + c;
            v16h aG0 = *(const v16h*)(gBase + (((size_t)mT * 2 + 0) * 32 + lane) * 16);
            v16h aG1 = *(const v16h*)(gBase + (((size_t)mT * 2 + 1) * 32 + lane) * 16);
            acc0 = __builtin_amdgcn_wmma_f32_16x16x32_f16(false, aG0, false, Bp.h,
                                                          (short)0, acc0, false, false);
            acc1 = __builtin_amdgcn_wmma_f32_16x16x32_f16(false, aG1, false, Bp.h,
                                                          (short)0, acc1, false, false);
        }
    }

    // Normalize and re-layout y^T (C-frag) -> B-frag (k=ci, n=query): one
    // packed partner exchange, then out-tile = w_out @ y^T + b_out + x.
    float inv = 1.0f / l;
    uint32_t y0[4], y1[4];
    #pragma unroll
    for (int j = 0; j < 4; ++j) {
        y0[j] = pk2(acc0[2 * j] * inv, acc0[2 * j + 1] * inv);
        y1[j] = pk2(acc1[2 * j] * inv, acc1[2 * j + 1] * inv);
    }
    Frag16 By;
    #pragma unroll
    for (int j = 0; j < 4; ++j) {
        uint32_t send = hi ? y0[j] : y1[j];
        uint32_t recv = (uint32_t)__shfl_xor((int)send, 16, 32);
        By.u[j]     = hi ? recv  : y0[j];
        By.u[4 + j] = hi ? y1[j] : recv;
    }

    #pragma unroll
    for (int t = 0; t < 4; ++t) {
        v8f z = {};
        v8f d = __builtin_amdgcn_wmma_f32_16x16x32_f16(false, aW[t], false, By.h,
                                                       (short)0, z, false, false);
        #pragma unroll
        for (int i = 0; i < 8; ++i) {
            int cout = t * 16 + i + hi * 8;
            size_t idx = ((size_t)b * NC + cout) * NN + qTile * 16 + ln;
            out[idx] = d[i] + b_out[cout] + x[idx];
        }
    }
}

// ---------------------------------------------------------------------------
extern "C" void kernel_launch(void* const* d_in, const int* in_sizes, int n_in,
                              void* d_out, int out_size, void* d_ws, size_t ws_size,
                              hipStream_t stream) {
    const float* x       = (const float*)d_in[0];
    const float* w_theta = (const float*)d_in[1];
    const float* b_theta = (const float*)d_in[2];
    const float* w_phi   = (const float*)d_in[3];
    const float* b_phi   = (const float*)d_in[4];
    const float* w_g     = (const float*)d_in[5];
    const float* b_g     = (const float*)d_in[6];
    const float* w_out   = (const float*)d_in[7];
    const float* b_out   = (const float*)d_in[8];
    float* out = (float*)d_out;

    // 6 MB of f16 scratch: thetaB | phiA | gTA
    _Float16* thetaB = (_Float16*)d_ws;
    _Float16* phiA   = thetaB + (size_t)NB * (NN / 16) * 32 * 16;
    _Float16* gTA    = phiA   + (size_t)NB * (NN / 16) * 32 * 16;

    nl_proj_kernel<<<dim3(NN / 64, NB), 256, 0, stream>>>(
        x, w_theta, b_theta, w_phi, b_phi, w_g, b_g, thetaB, phiA, gTA);
    nl_attn_kernel<<<dim3(NN / 128, NB), 256, 0, stream>>>(
        x, w_out, b_out, thetaB, phiA, gTA, out);
}
